// StructureUpdate_35132832481480
// MI455X (gfx1250) — matrix operand order repaired and checked by reading, not compiled, not run on hardware
//
#include <hip/hip_runtime.h>
#include <hip/hip_bf16.h>
#include <math.h>

typedef __attribute__((ext_vector_type(16))) _Float16 v16h;
typedef __attribute__((ext_vector_type(8)))  float    v8f;
typedef __attribute__((ext_vector_type(4)))  _Float16 v4h;

#define NRES 640
#define CS   256
#define CZ   128
#define CH   128
#define NH   8
#define PQn  8
#define PVn  12

// ---------------- WMMA helpers (gfx1250, wave32, 16x16x32 f16) ----------------

static __device__ __forceinline__ v8f wmma16(v16h a, v16h b, v8f c) {
  return __builtin_amdgcn_wmma_f32_16x16x32_f16(false, a, false, b, (short)0, c,
                                                false, false);
}

// A fragment: 16x32 tile in LDS (f16, row-major ld=32). Lane holds M=lane%16,
// K = two contiguous 8-runs: [kb..kb+8) and [16+kb..16+kb+8), kb = 8*(lane/16).
static __device__ __forceinline__ v16h load_a_frag(const _Float16* A, int lane) {
  v16h out;
  int m  = lane & 15;
  int kb = (lane >> 4) << 3;
#pragma unroll
  for (int e = 0; e < 16; ++e) {
    int k = ((e < 8) ? 0 : 16) + kb + (e & 7);
    out[e] = A[m * 32 + k];
  }
  return out;
}

// Same fragment but source tile is f32 in LDS (convert while loading).
static __device__ __forceinline__ v16h load_a_frag_f32(const float* A, int lane) {
  v16h out;
  int m  = lane & 15;
  int kb = (lane >> 4) << 3;
#pragma unroll
  for (int e = 0; e < 16; ++e) {
    int k = ((e < 8) ? 0 : 16) + kb + (e & 7);
    out[e] = (_Float16)A[m * 32 + k];
  }
  return out;
}

// B fragment: 32xN tile in LDS (f16, row-major ldb). Lane holds N=col0+lane%16,
// K = two contiguous 8-runs: [kb..kb+8) and [16+kb..16+kb+8), kb = 8*(lane/16).
static __device__ __forceinline__ v16h load_b_frag(const _Float16* Bt, int ldb,
                                                   int col0, int lane) {
  v16h out;
  int n  = col0 + (lane & 15);
  int kb = (lane >> 4) << 3;
#pragma unroll
  for (int e = 0; e < 16; ++e) out[e] = Bt[(((e < 8) ? 0 : 16) + kb + (e & 7)) * ldb + n];
  return out;
}

static __device__ __forceinline__ v4h cvt4(float4 v) {
  v4h h;
  h[0] = (_Float16)v.x; h[1] = (_Float16)v.y;
  h[2] = (_Float16)v.z; h[3] = (_Float16)v.w;
  return h;
}

// -------- CDNA5 async global->LDS copy (ASYNCcnt) -----------------------------
static __device__ __forceinline__ void async_copy16(unsigned lds_off,
                                                    const float* gptr) {
  asm volatile("global_load_async_to_lds_b128 %0, %1, off"
               :: "v"(lds_off), "v"((unsigned long long)(size_t)gptr)
               : "memory");
}
static __device__ __forceinline__ void wait_async0() {
  asm volatile("s_wait_asynccnt 0" ::: "memory");
}

// ---------------- Generic GEMM: C = act(A[MxK] @ W[KxN] + bias (+res)) -------
// block = 128 threads (4 waves); block tile: 16 rows x 64 cols. K % 32 == 0.
__global__ void gemm_kernel(const float* __restrict__ A, const float* __restrict__ W,
                            const float* __restrict__ bias,
                            const float* __restrict__ residual,
                            float* __restrict__ C, int M, int K, int Nc, int relu) {
  __shared__ __align__(16) _Float16 Alds[16 * 32];
  __shared__ __align__(16) _Float16 Blds[32 * 64];
  int cb = blockIdx.x, mt = blockIdx.y;
  int t = threadIdx.x, w = t >> 5, lane = t & 31;
  int m0 = mt * 16;
  bool fullN = (cb * 64 + 64) <= Nc;  // uniform over block
  int ra = t >> 3, ca4 = (t & 7) * 4; // A-stage mapping
  v8f acc = {};
  for (int k0 = 0; k0 < K; k0 += 32) {
    __syncthreads();
    // A tile 16x32, vectorized
    {
      float4 av = *(const float4*)(A + (size_t)(m0 + ra) * K + k0 + ca4);
      *(v4h*)(&Alds[ra * 32 + ca4]) = cvt4(av);
    }
    // B tile 32x64
    if (fullN) {
#pragma unroll
      for (int g = 0; g < 4; ++g) {
        int row = (t >> 4) + g * 8, col4 = (t & 15) * 4;
        float4 bv = *(const float4*)(W + (size_t)(k0 + row) * Nc + cb * 64 + col4);
        *(v4h*)(&Blds[row * 64 + col4]) = cvt4(bv);
      }
    } else {
#pragma unroll
      for (int e = t; e < 2048; e += 128) {
        int r = e >> 6, c = e & 63;
        int n = cb * 64 + c;
        int nc = (n < Nc) ? n : (Nc - 1);
        float v = W[(size_t)(k0 + r) * Nc + nc];
        Blds[e] = (n < Nc) ? (_Float16)v : (_Float16)0.f;
      }
    }
    __syncthreads();
    acc = wmma16(load_a_frag(Alds, lane), load_b_frag(Blds, 64, w * 16, lane), acc);
  }
  int n = cb * 64 + w * 16 + (lane & 15);
  if (n < Nc) {
#pragma unroll
    for (int r = 0; r < 8; ++r) {
      int m = m0 + r + ((lane >> 4) << 3);
      float v = acc[r];
      if (bias) v += bias[n];
      if (residual) v += residual[(size_t)m * Nc + n];
      if (relu) v = fmaxf(v, 0.f);
      C[(size_t)m * Nc + n] = v;
    }
  }
}

// ---------------- prep: rotation matrices + softplus head weights ------------
__global__ void prep_kernel(const float* __restrict__ quats,
                            const float* __restrict__ headw,
                            float* __restrict__ rot, float* __restrict__ hw) {
  int t = blockIdx.x * blockDim.x + threadIdx.x;
  if (t < NRES) {
    float w = quats[t * 4 + 0], x = quats[t * 4 + 1];
    float y = quats[t * 4 + 2], z = quats[t * 4 + 3];
    float* R = rot + t * 9;
    R[0] = w * w + x * x - y * y - z * z; R[1] = 2.f * (x * y - w * z); R[2] = 2.f * (x * z + w * y);
    R[3] = 2.f * (x * y + w * z); R[4] = w * w - x * x + y * y - z * z; R[5] = 2.f * (y * z - w * x);
    R[6] = 2.f * (x * z - w * y); R[7] = 2.f * (y * z + w * x); R[8] = w * w - x * x - y * y + z * z;
  }
  if (t < NH) {
    float xw = headw[t];
    float sp = (xw > 20.f) ? xw : log1pf(__expf(xw));
    hw[t] = sp * 0.09622504486493763f;  // sqrt(1/108)
  }
}

// ---------------- apply frame to raw point projections -----------------------
__global__ void pts_kernel(const float* __restrict__ raw, const float* __restrict__ rot,
                           const float* __restrict__ trans, float* __restrict__ out, int HP) {
  int idx = blockIdx.x * blockDim.x + threadIdx.x;
  if (idx >= NRES * HP) return;
  int n = idx / HP, hp = idx % HP;
  float v0 = raw[((size_t)n * 3 + 0) * HP + hp];
  float v1 = raw[((size_t)n * 3 + 1) * HP + hp];
  float v2 = raw[((size_t)n * 3 + 2) * HP + hp];
  const float* R = rot + n * 9;
#pragma unroll
  for (int i = 0; i < 3; ++i)
    out[(size_t)idx * 3 + i] =
        R[i * 3 + 0] * v0 + R[i * 3 + 1] * v1 + R[i * 3 + 2] * v2 + trans[n * 3 + i];
}

__global__ void sumsq_kernel(const float* __restrict__ pts, float* __restrict__ out,
                             int stride) {
  int t = blockIdx.x * blockDim.x + threadIdx.x;
  if (t >= NRES * NH) return;
  const float* p = pts + (size_t)t * stride;
  float s = 0.f;
#pragma unroll
  for (int e = 0; e < 24; ++e) { float v = p[e]; s += v * v; }
  out[t] = s;
}

// ---------------- fused z-projection: b_pair (8) + zdz (32) ------------------
// One block per i. z tiles streamed to LDS with the CDNA5 async engine,
// double-buffered: WMMA on chunk c overlaps async fill of chunk c+1.
__global__ void zproj_kernel(const float* __restrict__ z, const float* __restrict__ wb,
                             const float* __restrict__ bb, const float* __restrict__ wdz,
                             const float* __restrict__ bdz, float* __restrict__ bpair,
                             _Float16* __restrict__ zdz) {
  __shared__ __align__(16) _Float16 Wlds[128 * 48];
  __shared__ __align__(16) float Az[2][16 * 32];
  int i = blockIdx.x;
  int t = threadIdx.x, w = t >> 5, lane = t & 31;
  // stage W (128x48) once: col<8 -> wb, col<40 -> wdz, else 0 (clamped loads)
  for (int e = t; e < 128 * 48; e += 128) {
    int k = e / 48, c = e % 48;
    int c8 = (c < 8) ? c : 0;
    int c32 = (c >= 8 && c < 40) ? (c - 8) : 0;
    float v = (c < 8) ? wb[k * 8 + c8] : ((c < 40) ? wdz[k * 32 + c32] : 0.f);
    Wlds[e] = (_Float16)v;
  }
  int ra = t >> 3, ca4 = (t & 7) * 4;  // each thread moves 16B of the z tile
  unsigned lds0 = (unsigned)(size_t)&Az[0][ra * 32 + ca4];
  unsigned lds1 = (unsigned)(size_t)&Az[1][ra * 32 + ca4];
  __syncthreads();
  for (int jt = 0; jt < 40; ++jt) {
    const float* zrow = z + ((size_t)i * NRES + jt * 16 + ra) * CZ + ca4;
    async_copy16(lds0, zrow);  // chunk 0 (k0 = 0)
    if (jt + 1 < 40) __builtin_prefetch(zrow + 16 * CZ, 0, 1);
    v8f acc = {};
#pragma unroll
    for (int c = 0; c < 4; ++c) {
      wait_async0();
      __syncthreads();
      if (c < 3) async_copy16(((c + 1) & 1) ? lds1 : lds0, zrow + (c + 1) * 32);
      acc = wmma16(load_a_frag_f32(Az[c & 1], lane),
                   load_b_frag(Wlds + c * 32 * 48, 48, w * 16, lane), acc);
    }
    int cc = w * 16 + (lane & 15);
    if (cc < 40) {
#pragma unroll
      for (int r = 0; r < 8; ++r) {
        int jj = jt * 16 + r + ((lane >> 4) << 3);
        size_t pr = (size_t)i * NRES + jj;
        if (cc < 8)
          bpair[pr * 8 + cc] = acc[r] + bb[cc];
        else
          zdz[pr * 32 + (cc - 8)] = (_Float16)(acc[r] + bdz[cc - 8]);
      }
    }
    __syncthreads();
  }
}

// ---------------- attention logits -------------------------------------------
__global__ void logits_kernel(const float* __restrict__ q, const float* __restrict__ kv,
                              const float* __restrict__ qpts, const float* __restrict__ kvpts,
                              const float* __restrict__ bpair, const float* __restrict__ Sq,
                              const float* __restrict__ Sk, const float* __restrict__ hw,
                              const float* __restrict__ mask, float* __restrict__ a) {
  __shared__ __align__(16) _Float16 Alds[16 * 32];
  __shared__ __align__(16) _Float16 Blds[32 * 64];
  int jb = blockIdx.x, it = blockIdx.y, h = blockIdx.z;
  int t = threadIdx.x, w = t >> 5, lane = t & 31;
  int i0 = it * 16, j0 = jb * 64;
  v8f accqk = {}, accpt = {};
  // point cross-term (K=24 padded to 32), single WMMA. Clamped scalar staging.
#pragma unroll
  for (int e = t; e < 512; e += 128) {
    int r = e >> 5, c = e & 31;
    int cc = (c < 24) ? c : 0;
    float v = qpts[((size_t)(i0 + r) * NH + h) * 24 + cc];
    Alds[e] = (c < 24) ? (_Float16)v : (_Float16)0.f;
  }
#pragma unroll
  for (int e = t; e < 2048; e += 128) {
    int kk = e >> 6, jj = e & 63;
    int kc = (kk < 24) ? kk : 0;
    float v = kvpts[((size_t)(j0 + jj) * NH + h) * 60 + kc];
    Blds[e] = (kk < 24) ? (_Float16)v : (_Float16)0.f;
  }
  __syncthreads();
  accpt = wmma16(load_a_frag(Alds, lane), load_b_frag(Blds, 64, w * 16, lane), accpt);
  // q . k over CH=128 (vectorized staging; B staged along k then scattered)
  int ra = t >> 3, ca4 = (t & 7) * 4;
  for (int k0 = 0; k0 < CH; k0 += 32) {
    __syncthreads();
    {
      float4 av = *(const float4*)(q + (size_t)(i0 + ra) * (NH * CH) + h * CH + k0 + ca4);
      *(v4h*)(&Alds[ra * 32 + ca4]) = cvt4(av);
    }
#pragma unroll
    for (int g = 0; g < 4; ++g) {
      int jj = (t >> 3) + g * 16, k4 = (t & 7) * 4;
      float4 bv = *(const float4*)(kv + (size_t)(j0 + jj) * (NH * 2 * CH) + h * 2 * CH + k0 + k4);
      Blds[(k4 + 0) * 64 + jj] = (_Float16)bv.x;
      Blds[(k4 + 1) * 64 + jj] = (_Float16)bv.y;
      Blds[(k4 + 2) * 64 + jj] = (_Float16)bv.z;
      Blds[(k4 + 3) * 64 + jj] = (_Float16)bv.w;
    }
    __syncthreads();
    accqk = wmma16(load_a_frag(Alds, lane), load_b_frag(Blds, 64, w * 16, lane), accqk);
  }
  float hwv = hw[h];
  int jj = j0 + w * 16 + (lane & 15);
#pragma unroll
  for (int r = 0; r < 8; ++r) {
    int ii = i0 + r + ((lane >> 4) << 3);
    float logit = 0.051031036307982884f * accqk[r]                       // sqrt(1/384)
                + 0.5773502691896258f * bpair[((size_t)ii * NRES + jj) * 8 + h]
                - 0.5f * hwv * (Sq[ii * NH + h] + Sk[jj * NH + h] - 2.f * accpt[r])
                + 100000.f * (mask[ii] * mask[jj] - 1.f);
    a[((size_t)h * NRES + ii) * NRES + jj] = logit;
  }
}

// ---------------- row softmax -> f16 probabilities ---------------------------
__global__ void softmax_kernel(const float* __restrict__ a, _Float16* __restrict__ ah) {
  __shared__ float red[128];
  int row = blockIdx.x;  // h*640 + i
  const float* p = a + (size_t)row * NRES;
  int t = threadIdx.x;
  float vals[5];
  float mx = -1e30f;
#pragma unroll
  for (int e = 0; e < 5; ++e) { vals[e] = p[t + e * 128]; mx = fmaxf(mx, vals[e]); }
  red[t] = mx; __syncthreads();
  for (int s = 64; s > 0; s >>= 1) { if (t < s) red[t] = fmaxf(red[t], red[t + s]); __syncthreads(); }
  mx = red[0]; __syncthreads();
  float sum = 0.f;
#pragma unroll
  for (int e = 0; e < 5; ++e) { vals[e] = __expf(vals[e] - mx); sum += vals[e]; }
  red[t] = sum; __syncthreads();
  for (int s = 64; s > 0; s >>= 1) { if (t < s) red[t] += red[t + s]; __syncthreads(); }
  float inv = 1.f / red[0];
#pragma unroll
  for (int e = 0; e < 5; ++e)
    ah[(size_t)row * NRES + t + e * 128] = (_Float16)(vals[e] * inv);
}

// ---------------- a @ V (shared for v and v_pts) -----------------------------
__global__ void attn_av_kernel(const _Float16* __restrict__ ah,
                               const float* __restrict__ Bsrc, float* __restrict__ Out,
                               int Nc, int bRowStride, int bColPerHead, int bColBase,
                               int outRowStride, int outColPerHead) {
  __shared__ __align__(16) _Float16 Alds[16 * 32];
  __shared__ __align__(16) _Float16 Blds[32 * 64];
  int cb = blockIdx.x, it = blockIdx.y, h = blockIdx.z;
  int t = threadIdx.x, w = t >> 5, lane = t & 31;
  int i0 = it * 16;
  bool fullN = (cb * 64 + 64) <= Nc;
  int ra = t >> 3, ca4 = (t & 7) * 4;
  v8f acc = {};
  for (int j0 = 0; j0 < NRES; j0 += 32) {
    __syncthreads();
    // A: probabilities, already f16
    *(v4h*)(&Alds[ra * 32 + ca4]) =
        *(const v4h*)(ah + ((size_t)h * NRES + i0 + ra) * NRES + j0 + ca4);
    if (fullN) {
#pragma unroll
      for (int g = 0; g < 4; ++g) {
        int row = (t >> 4) + g * 8, col4 = (t & 15) * 4;
        float4 bv = *(const float4*)(Bsrc + (size_t)(j0 + row) * bRowStride +
                                     h * bColPerHead + bColBase + cb * 64 + col4);
        *(v4h*)(&Blds[row * 64 + col4]) = cvt4(bv);
      }
    } else {
#pragma unroll
      for (int e = t; e < 2048; e += 128) {
        int kk = e >> 6, cc = e & 63;
        int cg = cb * 64 + cc;
        int cgc = (cg < Nc) ? cg : (Nc - 1);
        float v = Bsrc[(size_t)(j0 + kk) * bRowStride + h * bColPerHead + bColBase + cgc];
        Blds[e] = (cg < Nc) ? (_Float16)v : (_Float16)0.f;
      }
    }
    __syncthreads();
    acc = wmma16(load_a_frag(Alds, lane), load_b_frag(Blds, 64, w * 16, lane), acc);
  }
  int cg = cb * 64 + w * 16 + (lane & 15);
  if (cg < Nc) {
#pragma unroll
    for (int r = 0; r < 8; ++r) {
      int ii = i0 + r + ((lane >> 4) << 3);
      Out[(size_t)ii * outRowStride + h * outColPerHead + cg] = acc[r];
    }
  }
}

// ---------------- o_pair: per-i (8x640)@(640x32) -----------------------------
__global__ void opair_kernel(const _Float16* __restrict__ ah,
                             const _Float16* __restrict__ zdz, float* __restrict__ opair) {
  __shared__ __align__(16) _Float16 Alds[16 * 32];
  __shared__ __align__(16) _Float16 Blds[32 * 32];
  int i = blockIdx.x;
  int t = threadIdx.x, w = t >> 5, lane = t & 31;  // 64 threads, 2 waves
  int ra = t >> 2, ca8 = (t & 3) * 8;   // A: 8 halves per thread
  int rb = t >> 1, cb16 = (t & 1) * 16; // B: 16 halves per thread
  v8f acc = {};
  for (int j0 = 0; j0 < NRES; j0 += 32) {
    __syncthreads();
    if (ra < NH) {
      const _Float16* src = ah + ((size_t)ra * NRES + i) * NRES + j0 + ca8;
      *(v4h*)(&Alds[ra * 32 + ca8 + 0]) = *(const v4h*)(src + 0);
      *(v4h*)(&Alds[ra * 32 + ca8 + 4]) = *(const v4h*)(src + 4);
    } else {
      v4h zz = {};
      *(v4h*)(&Alds[ra * 32 + ca8 + 0]) = zz;
      *(v4h*)(&Alds[ra * 32 + ca8 + 4]) = zz;
    }
    {
      const _Float16* src = zdz + ((size_t)i * NRES + j0 + rb) * 32 + cb16;
#pragma unroll
      for (int u = 0; u < 4; ++u)
        *(v4h*)(&Blds[rb * 32 + cb16 + u * 4]) = *(const v4h*)(src + u * 4);
    }
    __syncthreads();
    acc = wmma16(load_a_frag(Alds, lane), load_b_frag(Blds, 32, w * 16, lane), acc);
  }
  int d = w * 16 + (lane & 15);
#pragma unroll
  for (int r = 0; r < 8; ++r) {
    int h = r + ((lane >> 4) << 3);
    if (h < NH) opair[((size_t)i * NH + h) * 32 + d] = acc[r];
  }
}

// ---------------- assemble feats (inverse frame + norms) ---------------------
__global__ void feats_kernel(const float* __restrict__ o, const float* __restrict__ optg,
                             const float* __restrict__ opair, const float* __restrict__ rot,
                             const float* __restrict__ trans, float* __restrict__ feats) {
  int n = blockIdx.x, t = threadIdx.x;
  float* f = feats + (size_t)n * 1664;
  for (int c = t; c < 1024; c += 256) f[c] = o[(size_t)n * 1024 + c];
  if (t < 96) {
    float g0 = optg[((size_t)n * 96 + t) * 3 + 0] - trans[n * 3 + 0];
    float g1 = optg[((size_t)n * 96 + t) * 3 + 1] - trans[n * 3 + 1];
    float g2 = optg[((size_t)n * 96 + t) * 3 + 2] - trans[n * 3 + 2];
    const float* R = rot + n * 9;
    float l0 = R[0] * g0 + R[3] * g1 + R[6] * g2;  // R^T
    float l1 = R[1] * g0 + R[4] * g1 + R[7] * g2;
    float l2 = R[2] * g0 + R[5] * g1 + R[8] * g2;
    float nrm = sqrtf(l0 * l0 + l1 * l1 + l2 * l2 + 1e-8f);
    f[1024 + t] = l0; f[1120 + t] = l1; f[1216 + t] = l2; f[1312 + t] = nrm;
  }
  for (int c = t; c < 256; c += 256) f[1408 + c] = opair[(size_t)n * 256 + c];
}

// ---------------- layer norm (C=256) -----------------------------------------
__global__ void ln_kernel(const float* __restrict__ x, const float* __restrict__ g,
                          const float* __restrict__ b, float* __restrict__ out) {
  __shared__ float red[256];
  int n = blockIdx.x, t = threadIdx.x;
  float v = x[(size_t)n * CS + t];
  red[t] = v; __syncthreads();
  for (int s = 128; s > 0; s >>= 1) { if (t < s) red[t] += red[t + s]; __syncthreads(); }
  float mu = red[0] / 256.f; __syncthreads();
  float d = v - mu;
  red[t] = d * d; __syncthreads();
  for (int s = 128; s > 0; s >>= 1) { if (t < s) red[t] += red[t + s]; __syncthreads(); }
  float r = rsqrtf(red[0] / 256.f + 1e-5f);
  out[(size_t)n * CS + t] = d * r * g[t] + b[t];
}

// ---------------- quaternion / translation update ----------------------------
__global__ void pose_kernel(const float* __restrict__ upd, const float* __restrict__ quats,
                            const float* __restrict__ rot, const float* __restrict__ trans,
                            const float* __restrict__ mask, float* __restrict__ outq,
                            float* __restrict__ outt) {
  int n = blockIdx.x * blockDim.x + threadIdx.x;
  if (n >= NRES) return;
  float m = mask[n];
  float vx = upd[n * 6 + 0] * m, vy = upd[n * 6 + 1] * m, vz = upd[n * 6 + 2] * m;
  float tx = upd[n * 6 + 3], ty = upd[n * 6 + 4], tz = upd[n * 6 + 5];
  float qw = quats[n * 4 + 0], qx = quats[n * 4 + 1];
  float qy = quats[n * 4 + 2], qz = quats[n * 4 + 3];
  float dw = -qx * vx - qy * vy - qz * vz;
  float dx =  qw * vx + qy * vz - qz * vy;
  float dy =  qw * vy + qz * vx - qx * vz;
  float dz =  qw * vz + qx * vy - qy * vx;
  float nw = qw + dw, nx = qx + dx, ny = qy + dy, nz = qz + dz;
  float inv = rsqrtf(nw * nw + nx * nx + ny * ny + nz * nz);
  outq[n * 4 + 0] = nw * inv; outq[n * 4 + 1] = nx * inv;
  outq[n * 4 + 2] = ny * inv; outq[n * 4 + 3] = nz * inv;
  const float* R = rot + n * 9;
#pragma unroll
  for (int i = 0; i < 3; ++i)
    outt[n * 3 + i] = trans[n * 3 + i] +
        (R[i * 3 + 0] * tx + R[i * 3 + 1] * ty + R[i * 3 + 2] * tz) * m;
}

// ---------------- host ---------------------------------------------------------

extern "C" void kernel_launch(void* const* d_in, const int* in_sizes, int n_in,
                              void* d_out, int out_size, void* d_ws, size_t ws_size,
                              hipStream_t stream) {
  (void)in_sizes; (void)n_in; (void)out_size; (void)ws_size;
  const float* s      = (const float*)d_in[0];
  const float* z      = (const float*)d_in[1];
  const float* quats  = (const float*)d_in[2];
  const float* trans  = (const float*)d_in[3];
  const float* mask   = (const float*)d_in[4];
  const float* wq     = (const float*)d_in[5];
  const float* bq     = (const float*)d_in[6];
  const float* wkv    = (const float*)d_in[7];
  const float* bkv    = (const float*)d_in[8];
  const float* wqp    = (const float*)d_in[9];
  const float* bqp    = (const float*)d_in[10];
  const float* wkvp   = (const float*)d_in[11];
  const float* bkvp   = (const float*)d_in[12];
  const float* wb     = (const float*)d_in[13];
  const float* bb_lin = (const float*)d_in[14];
  const float* head_w = (const float*)d_in[15];
  const float* wdz    = (const float*)d_in[16];
  const float* bdz    = (const float*)d_in[17];
  const float* wout   = (const float*)d_in[18];
  const float* bout   = (const float*)d_in[19];
  const float* ln1_g  = (const float*)d_in[20];
  const float* ln1_b  = (const float*)d_in[21];
  const float* wt1    = (const float*)d_in[22];
  const float* bt1    = (const float*)d_in[23];
  const float* wt2    = (const float*)d_in[24];
  const float* bt2    = (const float*)d_in[25];
  const float* wt3    = (const float*)d_in[26];
  const float* bt3    = (const float*)d_in[27];
  const float* ln2_g  = (const float*)d_in[28];
  const float* ln2_b  = (const float*)d_in[29];
  const float* wbb    = (const float*)d_in[30];
  const float* bbb    = (const float*)d_in[31];

  float* out_s = (float*)d_out;                 // 640*256
  float* out_q = out_s + (size_t)NRES * CS;     // 640*4
  float* out_t = out_q + (size_t)NRES * 4;      // 640*3

  float* Wp = (float*)d_ws;
  size_t off = 0;
  auto alloc = [&](size_t nfl) { float* p = Wp + off; off += nfl; return p; };
  float*    qb     = alloc((size_t)NRES * 1024);
  float*    kvb    = alloc((size_t)NRES * 2048);
  float*    qpraw  = alloc((size_t)NRES * 192);
  float*    kvpraw = alloc((size_t)NRES * 480);
  float*    qpts   = alloc((size_t)NRES * NH * PQn * 3);
  float*    kvpts  = alloc((size_t)NRES * NH * 20 * 3);
  float*    rotb   = alloc((size_t)NRES * 9);
  float*    Sqb    = alloc((size_t)NRES * NH);
  float*    Skb    = alloc((size_t)NRES * NH);
  float*    hwb    = alloc(8);
  float*    bpair  = alloc((size_t)NRES * NRES * NH);
  _Float16* zdzh   = (_Float16*)alloc((size_t)NRES * NRES * 32 / 2);
  float*    alog   = alloc((size_t)NH * NRES * NRES);
  _Float16* ah     = (_Float16*)alloc((size_t)NH * NRES * NRES / 2);
  float*    obuf   = alloc((size_t)NRES * 1024);
  float*    optg   = alloc((size_t)NRES * 288);
  float*    opairb = alloc((size_t)NRES * 256);
  float*    featsb = alloc((size_t)NRES * 1664);
  float*    s1     = alloc((size_t)NRES * CS);
  float*    sln1   = alloc((size_t)NRES * CS);
  float*    h1     = alloc((size_t)NRES * CS);
  float*    h2     = alloc((size_t)NRES * CS);
  float*    s2     = alloc((size_t)NRES * CS);
  float*    updb   = alloc((size_t)NRES * 6);

  prep_kernel<<<3, 256, 0, stream>>>(quats, head_w, rotb, hwb);

  // s projections
  gemm_kernel<<<dim3(16, 40), 128, 0, stream>>>(s, wq,   bq,   nullptr, qb,     NRES, CS, 1024, 0);
  gemm_kernel<<<dim3(32, 40), 128, 0, stream>>>(s, wkv,  bkv,  nullptr, kvb,    NRES, CS, 2048, 0);
  gemm_kernel<<<dim3(3,  40), 128, 0, stream>>>(s, wqp,  bqp,  nullptr, qpraw,  NRES, CS, 192,  0);
  gemm_kernel<<<dim3(8,  40), 128, 0, stream>>>(s, wkvp, bkvp, nullptr, kvpraw, NRES, CS, 480,  0);

  pts_kernel<<<(NRES * 64  + 255) / 256, 256, 0, stream>>>(qpraw,  rotb, trans, qpts,  64);
  pts_kernel<<<(NRES * 160 + 255) / 256, 256, 0, stream>>>(kvpraw, rotb, trans, kvpts, 160);
  sumsq_kernel<<<20, 256, 0, stream>>>(qpts,  Sqb, 24);
  sumsq_kernel<<<20, 256, 0, stream>>>(kvpts, Skb, 60);

  // single async-streamed pass over z: b_pair + zdz
  zproj_kernel<<<NRES, 128, 0, stream>>>(z, wb, bb_lin, wdz, bdz, bpair, zdzh);

  // logits + softmax
  logits_kernel<<<dim3(10, 40, 8), 128, 0, stream>>>(qb, kvb, qpts, kvpts, bpair,
                                                     Sqb, Skb, hwb, mask, alog);
  softmax_kernel<<<NH * NRES, 128, 0, stream>>>(alog, ah);

  // attention outputs
  attn_av_kernel<<<dim3(2, 40, 8), 128, 0, stream>>>(ah, kvb,   obuf, 128, 2048, 256, 128, 1024, 128);
  attn_av_kernel<<<dim3(1, 40, 8), 128, 0, stream>>>(ah, kvpts, optg, 36,  480,  60,  24,  288,  36);
  opair_kernel<<<NRES, 64, 0, stream>>>(ah, zdzh, opairb);

  feats_kernel<<<NRES, 256, 0, stream>>>(obuf, optg, opairb, rotb, trans, featsb);

  // output projection + transition
  gemm_kernel<<<dim3(4, 40), 128, 0, stream>>>(featsb, wout, bout, s, s1, NRES, 1664, CS, 0);
  ln_kernel<<<NRES, 256, 0, stream>>>(s1, ln1_g, ln1_b, sln1);
  gemm_kernel<<<dim3(4, 40), 128, 0, stream>>>(sln1, wt1, bt1, nullptr, h1, NRES, CS, CS, 1);
  gemm_kernel<<<dim3(4, 40), 128, 0, stream>>>(h1,   wt2, bt2, nullptr, h2, NRES, CS, CS, 1);
  gemm_kernel<<<dim3(4, 40), 128, 0, stream>>>(h2,   wt3, bt3, sln1,    s2, NRES, CS, CS, 0);
  ln_kernel<<<NRES, 256, 0, stream>>>(s2, ln2_g, ln2_b, out_s);

  // backbone update
  gemm_kernel<<<dim3(1, 40), 128, 0, stream>>>(out_s, wbb, bbb, nullptr, updb, NRES, CS, 6, 0);
  pose_kernel<<<5, 128, 0, stream>>>(updb, quats, rotb, trans, mask, out_q, out_t);
}
